// Maha_RA_56272661512683
// MI455X (gfx1250) — matrix operand order repaired and checked by reading, not compiled, not run on hardware
//
#include <hip/hip_runtime.h>
#include <hip/hip_bf16.h>
#include <math.h>

// ---------------------------------------------------------------------------
// Problem constants (from reference): B=16, C=256, H=W=64, L=40, TD=768,
// NC=8, EMB=(768+16)/2=392, M = B*H*W = 65536.
// ---------------------------------------------------------------------------
#define BB   16
#define CC   256
#define HH   64
#define MM   65536       // B*H*W
#define LL   40
#define TD   768
#define EMBD 392
#define NCH  8

typedef __attribute__((ext_vector_type(16))) __bf16       v16bf;
typedef __attribute__((ext_vector_type(8)))  float        v8f;
typedef __attribute__((ext_vector_type(4)))  unsigned int v4u;

union BfFrag { v16bf bf; v4u q[2]; };

// A fragment: per-lane K = halfbase+0..7 at +0, halfbase+16..23 at +16 elems
#define LOAD_AFRAG(dst, p, k0)                                   \
  do {                                                           \
    (dst).q[0] = *(const v4u*)((p) + (k0));                      \
    (dst).q[1] = *(const v4u*)((p) + (k0) + 16);                 \
  } while (0)

// B fragment: per-lane K = halfbase+0..7 at +0, halfbase+16..23 at +16 elems
#define LOAD_BFRAG(dst, p, k0)                                   \
  do {                                                           \
    (dst).q[0] = *(const v4u*)((p) + (k0));                      \
    (dst).q[1] = *(const v4u*)((p) + (k0) + 16);                 \
  } while (0)

#define ZERO_FRAG(dst)                                           \
  do {                                                           \
    v4u z_ = {0u, 0u, 0u, 0u};                                   \
    (dst).q[0] = z_; (dst).q[1] = z_;                            \
  } while (0)

#define WMMA_BF16(a, b, c)                                       \
  __builtin_amdgcn_wmma_f32_16x16x32_bf16(false, (a).bf, false, (b).bf, \
                                          (short)0, (c), false, false)

__device__ __forceinline__ unsigned short f2bf(float x) {
  unsigned u = __float_as_uint(x);
  unsigned r = u + 0x7FFFu + ((u >> 16) & 1u);     // round to nearest even
  if ((u & 0x7F800000u) == 0x7F800000u) r = u;     // keep inf/NaN
  return (unsigned short)(r >> 16);
}

__device__ __forceinline__ float sigmoidf_(float x) { return 1.f / (1.f + __expf(-x)); }

__device__ __forceinline__ float block_reduce_sum256(float v) {
  __shared__ float s[256];
  int t = threadIdx.x;
  s[t] = v; __syncthreads();
  for (int o = 128; o > 0; o >>= 1) {
    if (t < o) s[t] += s[t + o];
    __syncthreads();
  }
  float r = s[0];
  __syncthreads();
  return r;
}

// ---------------------------------------------------------------------------
// Layout converters
// ---------------------------------------------------------------------------

// img NCHW f32 -> bf16 [M, C] (channels-last, K contiguous for WMMA A)
__global__ void cvt_img_kernel(const float* __restrict__ img,
                               unsigned short* __restrict__ Aimg) {
  size_t idx = (size_t)blockIdx.x * blockDim.x + threadIdx.x;  // M*C threads
  int c = (int)(idx & 255);
  int m = (int)(idx >> 8);
  int b = m >> 12;
  int p = m & 4095;
  Aimg[idx] = f2bf(img[(((size_t)(b * CC + c)) << 12) + p]);
}

// f32 [O,I] -> bf16 [N=O, K=I] (same layout)
__global__ void cvt_w_kernel(const float* __restrict__ in,
                             unsigned short* __restrict__ out, int n) {
  int i = blockIdx.x * 256 + threadIdx.x;
  if (i < n) out[i] = f2bf(in[i]);
}

// Wf2 OIHW (256,256,3,3) f32 -> bf16 [9][O][I] (K=I contiguous per tap)
__global__ void cvt_wf2_kernel(const float* __restrict__ in,
                               unsigned short* __restrict__ out) {
  int idx = blockIdx.x * 256 + threadIdx.x;     // 256*256*9 = 589824
  int o = idx / (CC * 9);
  int rem = idx % (CC * 9);
  int i = rem / 9;
  int k = rem % 9;
  out[(size_t)k * (CC * CC) + (size_t)o * CC + i] = f2bf(in[idx]);
}

// ---------------------------------------------------------------------------
// WMMA GEMM with LDS-staged weight panel.
// Block = 256 threads = 8 waves: covers 128 M rows x one 64-col N panel.
// The 64x256 bf16 B panel (32 KB) is staged once in LDS cooperatively; each
// wave then computes a 16x64 tile: 4 accumulators, B fragments via
// ds_load_b128, A fragments double-buffered from global (2 loads/K-step).
// Live VGPRs ~70: no spills, high occupancy.
// EPI: 0 = +bias, f32 out (Wf3)
//      1 = *scale+bias, SiLU, bf16 out (Wf1)
//      2 = *scale+bias, SiLU, tanh, f32 out (map_visu)
// ---------------------------------------------------------------------------
template <int EPI>
__global__ __launch_bounds__(256) void gemm_lds_bf16(
    const unsigned short* __restrict__ A,
    const unsigned short* __restrict__ Bw,
    const float* __restrict__ scale,
    const float* __restrict__ bias,
    float* __restrict__ outF,
    unsigned short* __restrict__ outB) {
  __shared__ unsigned short sB[64 * CC];   // 32 KB weight panel [n][k]

  const int tid = threadIdx.x;
  const int ng = blockIdx.x & 3;           // N panel (N/64 = 4)
  const int mb = blockIdx.x >> 2;          // 128-row M block
  const int n0 = ng << 6;
  const int mbase = mb << 7;

  // cooperative stage: 64 rows x 256 K of bf16 = 2048 x 16B
  for (int i = tid; i < (64 * CC) / 8; i += 256) {
    const int n = i >> 5;                  // 32 chunks of 8 per row
    const int k8 = (i & 31) << 3;
    *(v4u*)(&sB[n * CC + k8]) = *(const v4u*)(Bw + (size_t)(n0 + n) * CC + k8);
  }
  __syncthreads();

  const int lane = tid & 31, widx = tid >> 5;
  const int row = lane & 15, half = lane >> 4;
  const int m0 = mbase + (widx << 4);

  v8f acc[4] = {};
  const unsigned short* pa  = A + (size_t)(m0 + row) * CC + half * 8;
  const unsigned short* sb0 = &sB[row * CC + half * 8];

  BfFrag a2[2];
  LOAD_AFRAG(a2[0], pa, 0);
#pragma unroll
  for (int kk = 0; kk < 8; ++kk) {
    const int k0 = kk << 5;
    const int cur = kk & 1, nxt = cur ^ 1;
    if (kk < 7) LOAD_AFRAG(a2[nxt], pa, k0 + 32);   // prefetch next K-step
#pragma unroll
    for (int t = 0; t < 4; ++t) {
      BfFrag fb;
      LOAD_BFRAG(fb, sb0 + t * 16 * CC, k0);        // ds_load_b128 x2
      acc[t] = WMMA_BF16(a2[cur], fb, acc[t]);
    }
  }

#pragma unroll
  for (int t = 0; t < 4; ++t) {
    const int n = n0 + (t << 4) + row;
    const float sc = (EPI >= 1) ? scale[n] : 1.f;
    const float bi = bias[n];
#pragma unroll
    for (int r = 0; r < 8; ++r) {
      const int m = m0 + r + (half << 3);
      float y = acc[t][r];
      if (EPI >= 1) { y = y * sc + bi; y = y * sigmoidf_(y); }
      else          { y += bi; }
      if (EPI == 2) y = tanhf(y);
      size_t o = (size_t)m * CC + n;
      if (EPI == 1) outB[o] = f2bf(y);
      else          outF[o] = y;
    }
  }
}

// ---------------------------------------------------------------------------
// 3x3 pad-1 conv 256->256: 9 accumulated WMMA GEMM taps with the tap's
// 64x256 weight panel staged in LDS per iteration. Barriers are uniform:
// out-of-bounds rows run with zeroed A fragments instead of skipping.
// Epilogue: *scale+bias, SiLU -> bf16.
// ---------------------------------------------------------------------------
__global__ __launch_bounds__(256) void conv3x3_lds(
    const unsigned short* __restrict__ A,
    const unsigned short* __restrict__ Wt,  // [9][O][I] bf16
    const float* __restrict__ scale,
    const float* __restrict__ bias,
    unsigned short* __restrict__ outB) {
  __shared__ unsigned short sB[64 * CC];   // 32 KB tap panel

  const int tid = threadIdx.x;
  const int ng = blockIdx.x & 3;
  const int mb = blockIdx.x >> 2;
  const int n0 = ng << 6;
  const int mbase = mb << 7;

  const int lane = tid & 31, widx = tid >> 5;
  const int row = lane & 15, half = lane >> 4;
  const int m0 = mbase + (widx << 4);
  const int b  = m0 >> 12;
  const int h  = (m0 >> 6) & 63;
  const int w0 = m0 & 63;                  // 0,16,32,48

  v8f acc[4] = {};
  for (int tap = 0; tap < 9; ++tap) {
    const int dy = tap / 3, dx = tap % 3;
    const unsigned short* wpanel = Wt + (size_t)tap * (CC * CC);

    // stage this tap's 64x256 panel
    for (int i = tid; i < (64 * CC) / 8; i += 256) {
      const int n = i >> 5;
      const int k8 = (i & 31) << 3;
      *(v4u*)(&sB[n * CC + k8]) =
          *(const v4u*)(wpanel + (size_t)(n0 + n) * CC + k8);
    }
    __syncthreads();
    if (tap < 8)                           // warm next tap's panel in L2
      __builtin_prefetch(Wt + (size_t)(tap + 1) * (CC * CC) +
                             (size_t)(n0) * CC + tid * 64, 0, 1);

    const int hs  = h + dy - 1;
    const int wsp = w0 + row + dx - 1;
    const bool valid = ((unsigned)hs < 64u) && ((unsigned)wsp < 64u);
    const unsigned short* pa =
        A + ((size_t)((b << 12) + ((valid ? hs : 0) << 6) + (valid ? wsp : 0))) * CC +
        half * 8;
    const unsigned short* sb0 = &sB[row * CC + half * 8];

    BfFrag a2[2];
    if (valid) LOAD_AFRAG(a2[0], pa, 0); else ZERO_FRAG(a2[0]);
#pragma unroll
    for (int kk = 0; kk < 8; ++kk) {
      const int k0 = kk << 5;
      const int cur = kk & 1, nxt = cur ^ 1;
      if (kk < 7) {
        if (valid) LOAD_AFRAG(a2[nxt], pa, k0 + 32); else ZERO_FRAG(a2[nxt]);
      }
#pragma unroll
      for (int t = 0; t < 4; ++t) {
        BfFrag fb;
        LOAD_BFRAG(fb, sb0 + t * 16 * CC, k0);
        acc[t] = WMMA_BF16(a2[cur], fb, acc[t]);
      }
    }
    __syncthreads();                        // panel reuse barrier
  }

#pragma unroll
  for (int t = 0; t < 4; ++t) {
    const int n = n0 + (t << 4) + row;
    const float sc = scale[n], bi = bias[n];
#pragma unroll
    for (int r = 0; r < 8; ++r) {
      const int m = m0 + r + (half << 3);
      float y = acc[t][r] * sc + bi;
      y = y * sigmoidf_(y);
      outB[(size_t)m * CC + n] = f2bf(y);
    }
  }
}

// ---------------------------------------------------------------------------
// Language path
// ---------------------------------------------------------------------------
// flang_avg[b, td] = mean of flang[b, 1..len-2, td]  (cumsum-mask simplifies)
__global__ void flang_avg_kernel(const float* __restrict__ flang,
                                 const int* __restrict__ mask,
                                 float* __restrict__ favg) {
  int b = blockIdx.x;
  __shared__ int s_len;
  if (threadIdx.x == 0) {
    int len = 0;
    for (int l = 0; l < LL; ++l) len += mask[b * LL + l];
    s_len = len;
  }
  __syncthreads();
  int len = s_len;
  float denom = (float)(len - 2);
  for (int td = threadIdx.x; td < TD; td += blockDim.x) {
    float s = 0.f;
    for (int l = 1; l <= len - 2; ++l) s += flang[((size_t)b * LL + l) * TD + td];
    favg[(size_t)b * TD + td] = s / denom;
  }
}

// emb[b,e] = favg[b] . W1[e] + b1[e]
__global__ void emb_kernel(const float* __restrict__ favg,
                           const float* __restrict__ W1,
                           const float* __restrict__ b1,
                           float* __restrict__ emb) {
  int b = blockIdx.x;
  for (int e = threadIdx.x; e < EMBD; e += blockDim.x) {
    float s = b1[e];
    for (int k = 0; k < TD; ++k) s += favg[(size_t)b * TD + k] * W1[(size_t)e * TD + k];
    emb[(size_t)b * EMBD + e] = s;
  }
}

// maps -> integer-snapped gaussian centers yc,xc [B,NC]
__global__ void maps_kernel(const float* __restrict__ emb,
                            const float* __restrict__ W2,
                            const float* __restrict__ b2,
                            float* __restrict__ yc, float* __restrict__ xc) {
  int t = threadIdx.x;          // 256 = 16 b * 16 o
  int b = t >> 4, o = t & 15;
  float s = b2[o];
  for (int k = 0; k < EMBD; ++k) s += emb[(size_t)b * EMBD + k] * W2[(size_t)o * EMBD + k];
  float v = sigmoidf_(s);
  float cc = (float)(int)(v * (float)HH);   // truncation like astype(int32)
  int nc = o >> 1;
  if (o & 1) xc[b * NCH + nc] = cc;
  else       yc[b * NCH + nc] = cc;
}

// ---------------------------------------------------------------------------
// sigma path: mean(conv3x3_valid(img, Ws)) reordered as window-sums
// ---------------------------------------------------------------------------
// S[b,i,k] = sum over 62x62 window (ky,kx)=k of img[b,i]
__global__ void winsum_kernel(const float* __restrict__ img,
                              float* __restrict__ S9) {
  int bi = blockIdx.x;          // b*256 + i
  float s[9] = {0.f};
  const float* p = img + ((size_t)bi << 12);
  for (int q = threadIdx.x; q < 4096; q += blockDim.x) {
    int h = q >> 6, w = q & 63;
    float v = p[q];
    for (int ky = 0; ky < 3; ++ky) {
      if ((unsigned)(h - ky) >= 62u) continue;
      for (int kx = 0; kx < 3; ++kx) {
        if ((unsigned)(w - kx) >= 62u) continue;
        s[ky * 3 + kx] += v;
      }
    }
  }
  for (int j = 0; j < 9; ++j) {
    float r = block_reduce_sum256(s[j]);
    if (threadIdx.x == 0) S9[(size_t)bi * 9 + j] = r;
  }
}

// sigw[b,o] = sigmoid( (sum_{i,k} S[b,i,k]*Ws[o,i,k]) / 3844 + bs[o] )
__global__ void sigma_kernel(const float* __restrict__ S9,
                             const float* __restrict__ Ws,
                             const float* __restrict__ bs,
                             float* __restrict__ sigw) {
  int b = blockIdx.x >> 3;
  int o = blockIdx.x & 7;
  float s = 0.f;
  for (int t = threadIdx.x; t < CC * 9; t += blockDim.x) {
    int i = t / 9, k = t % 9;
    s += S9[((size_t)b * CC + i) * 9 + k] * Ws[((size_t)o * CC + i) * 9 + k];
  }
  float r = block_reduce_sum256(s);
  if (threadIdx.x == 0)
    sigw[b * NCH + o] = sigmoidf_(r / 3844.f + bs[o]);
}

// gaussian weights [B,NC,H,W]
__global__ void gauss_kernel(const float* __restrict__ sigma_w,
                             const float* __restrict__ sigw,
                             const float* __restrict__ yc,
                             const float* __restrict__ xc,
                             float* __restrict__ gw) {
  int idx = blockIdx.x * 256 + threadIdx.x;    // 16*8*64*64
  int w = idx & 63, h = (idx >> 6) & 63;
  int nc = (idx >> 12) & 7, b = idx >> 15;
  float sx = sigma_w[0] * (float)HH;
  float sy = sx * sigw[b * NCH + nc];
  float dyv = (float)h - yc[b * NCH + nc];
  float dxv = (float)w - xc[b * NCH + nc];
  gw[idx] = __expf(-(dyv * dyv / (2.f * sy * sy) + dxv * dxv / (2.f * sx * sx)));
}

// gamma/beta (8-ch 1x1 convs) -> g = mean(gamma), bmax = max(beta), per pixel
__global__ void gb_kernel(const float* __restrict__ gw,
                          const float* __restrict__ Wg, const float* __restrict__ sg,
                          const float* __restrict__ bg,
                          const float* __restrict__ Wb, const float* __restrict__ sb,
                          const float* __restrict__ bb,
                          float* __restrict__ gmean, float* __restrict__ bmaxv) {
  int idx = blockIdx.x * 256 + threadIdx.x;    // M
  int b = idx >> 12, p = idx & 4095;
  float wv[NCH];
  for (int i = 0; i < NCH; ++i) wv[i] = gw[(((size_t)(b * NCH + i)) << 12) + p];
  float gs = 0.f, bm = -3.4e38f;
  for (int o = 0; o < NCH; ++o) {
    float ag = 0.f, ab = 0.f;
    for (int i = 0; i < NCH; ++i) {
      ag += wv[i] * Wg[o * NCH + i];
      ab += wv[i] * Wb[o * NCH + i];
    }
    float yg = ag * sg[o] + bg[o]; yg = yg * sigmoidf_(yg);
    float yb = ab * sb[o] + bb[o]; yb = yb * sigmoidf_(yb);
    gs += tanhf(yg);
    bm = fmaxf(bm, tanhf(yb));
  }
  gmean[idx] = gs * (1.f / NCH);
  bmaxv[idx] = bm;
}

// v = l2norm_c( g*map_visu + bmax )  -> bf16 [M,C]
__global__ void vpre_kernel(const float* __restrict__ mapvisu,
                            const float* __restrict__ gmean,
                            const float* __restrict__ bmaxv,
                            unsigned short* __restrict__ vpre) {
  int m = blockIdx.x;
  float v = gmean[m] * mapvisu[(size_t)m * CC + threadIdx.x] + bmaxv[m];
  float ss = block_reduce_sum256(v * v);
  float inv = 1.f / fmaxf(sqrtf(ss), 1e-12f);
  vpre[(size_t)m * CC + threadIdx.x] = f2bf(v * inv);
}

// out = l2norm_c( (1+aw)*img + (1-aw)*v_film ), written back in NCHW
__global__ void final_kernel(const float* __restrict__ vfilm,
                             const float* __restrict__ img,
                             const float* __restrict__ addw,
                             float* __restrict__ out) {
  int m = blockIdx.x;
  int b = m >> 12, p = m & 4095, c = threadIdx.x;
  float aw = tanhf(addw[0]);
  float vf = vfilm[(size_t)m * CC + c];
  float iv = img[(((size_t)(b * CC + c)) << 12) + p];
  float va = (1.f + aw) * iv + (1.f - aw) * vf;
  float ss = block_reduce_sum256(va * va);
  float inv = 1.f / fmaxf(sqrtf(ss), 1e-12f);
  out[(((size_t)(b * CC + c)) << 12) + p] = va * inv;
}

// ---------------------------------------------------------------------------
// Workspace layout (bytes)
// ---------------------------------------------------------------------------
#define OFF_AIMG    0ull                                 // bf16 [M,C]  (reused as ACT1)
#define OFF_MAPV    (OFF_AIMG + 33554432ull)             // f32  [M,C]  (reused as VFILM)
#define OFF_VPRE    (OFF_MAPV + 67108864ull)             // bf16 [M,C]
#define OFF_ACT2    (OFF_VPRE + 33554432ull)             // bf16 [M,C]
#define OFF_WV      (OFF_ACT2 + 33554432ull)             // bf16 256*256
#define OFF_WF1     (OFF_WV   + 131072ull)
#define OFF_WF3     (OFF_WF1  + 131072ull)
#define OFF_WF2T    (OFF_WF3  + 131072ull)               // bf16 9*256*256
#define OFF_S9      (OFF_WF2T + 1179648ull)              // f32 16*256*9
#define OFF_SIGW    (OFF_S9   + 147456ull)               // f32 16*8
#define OFF_FAVG    (OFF_SIGW + 512ull)                  // f32 16*768
#define OFF_EMB     (OFF_FAVG + 49152ull)                // f32 16*392
#define OFF_YC      (OFF_EMB  + 25088ull)                // f32 16*8
#define OFF_XC      (OFF_YC   + 512ull)                  // f32 16*8
#define OFF_GW      (OFF_XC   + 512ull)                  // f32 16*8*64*64
#define OFF_G       (OFF_GW   + 2097152ull)              // f32 [M]
#define OFF_BMAX    (OFF_G    + 262144ull)               // f32 [M]

extern "C" void kernel_launch(void* const* d_in, const int* in_sizes, int n_in,
                              void* d_out, int out_size, void* d_ws, size_t ws_size,
                              hipStream_t stream) {
  (void)in_sizes; (void)n_in; (void)out_size; (void)ws_size;

  const float* img     = (const float*)d_in[0];
  const float* flang   = (const float*)d_in[1];
  const int*   wmask   = (const int*)  d_in[2];
  const float* sigma_w = (const float*)d_in[3];
  const float* W1      = (const float*)d_in[4];
  const float* b1      = (const float*)d_in[5];
  const float* W2      = (const float*)d_in[6];
  const float* b2      = (const float*)d_in[7];
  const float* Wv      = (const float*)d_in[8];
  const float* sv      = (const float*)d_in[9];
  const float* bv      = (const float*)d_in[10];
  const float* Wg      = (const float*)d_in[11];
  const float* sg      = (const float*)d_in[12];
  const float* bg      = (const float*)d_in[13];
  const float* Wb      = (const float*)d_in[14];
  const float* sb      = (const float*)d_in[15];
  const float* bbv     = (const float*)d_in[16];
  const float* Wf1     = (const float*)d_in[17];
  const float* sf1     = (const float*)d_in[18];
  const float* bf1     = (const float*)d_in[19];
  const float* Wf2     = (const float*)d_in[20];
  const float* sf2     = (const float*)d_in[21];
  const float* bf2     = (const float*)d_in[22];
  const float* Wf3     = (const float*)d_in[23];
  const float* bf3     = (const float*)d_in[24];
  const float* Ws      = (const float*)d_in[25];
  const float* bs      = (const float*)d_in[26];
  const float* addw    = (const float*)d_in[27];
  float* out = (float*)d_out;

  char* ws = (char*)d_ws;
  unsigned short* Aimg  = (unsigned short*)(ws + OFF_AIMG);  // then ACT1
  float*          mapv  = (float*)(ws + OFF_MAPV);           // then VFILM
  unsigned short* vpre  = (unsigned short*)(ws + OFF_VPRE);
  unsigned short* act2  = (unsigned short*)(ws + OFF_ACT2);
  unsigned short* wv16  = (unsigned short*)(ws + OFF_WV);
  unsigned short* wf116 = (unsigned short*)(ws + OFF_WF1);
  unsigned short* wf316 = (unsigned short*)(ws + OFF_WF3);
  unsigned short* wf2t  = (unsigned short*)(ws + OFF_WF2T);
  float* S9    = (float*)(ws + OFF_S9);
  float* sigw  = (float*)(ws + OFF_SIGW);
  float* favg  = (float*)(ws + OFF_FAVG);
  float* embp  = (float*)(ws + OFF_EMB);
  float* yc    = (float*)(ws + OFF_YC);
  float* xc    = (float*)(ws + OFF_XC);
  float* gw    = (float*)(ws + OFF_GW);
  float* gmean = (float*)(ws + OFF_G);
  float* bmaxv = (float*)(ws + OFF_BMAX);

  const dim3 blk(256);

  // --- layout conversions ---
  cvt_img_kernel<<<MM, blk, 0, stream>>>(img, Aimg);
  cvt_w_kernel<<<256, blk, 0, stream>>>(Wv,  wv16,  CC * CC);
  cvt_w_kernel<<<256, blk, 0, stream>>>(Wf1, wf116, CC * CC);
  cvt_w_kernel<<<256, blk, 0, stream>>>(Wf3, wf316, CC * CC);
  cvt_wf2_kernel<<<2304, blk, 0, stream>>>(Wf2, wf2t);

  // --- language path ---
  flang_avg_kernel<<<BB, blk, 0, stream>>>(flang, wmask, favg);
  emb_kernel<<<BB, blk, 0, stream>>>(favg, W1, b1, embp);
  maps_kernel<<<1, blk, 0, stream>>>(embp, W2, b2, yc, xc);

  // --- sigma path ---
  winsum_kernel<<<BB * CC, blk, 0, stream>>>(img, S9);
  sigma_kernel<<<BB * NCH, blk, 0, stream>>>(S9, Ws, bs, sigw);

  // --- gaussian + gamma/beta ---
  gauss_kernel<<<(BB * NCH * HH * HH) / 256, blk, 0, stream>>>(sigma_w, sigw, yc, xc, gw);
  gb_kernel<<<MM / 256, blk, 0, stream>>>(gw, Wg, sg, bg, Wb, sb, bbv, gmean, bmaxv);

  // --- map_visu = tanh(silu(img*Wv*sv+bv)) : WMMA GEMM, f32 out ---
  // blocks = (M/128) * (N/64) = 512 * 4 = 2048
  gemm_lds_bf16<2><<<2048, blk, 0, stream>>>(Aimg, wv16, sv, bv, mapv, nullptr);

  // --- v = l2norm(g*map_visu + bmax) -> bf16 ---
  vpre_kernel<<<MM, blk, 0, stream>>>(mapv, gmean, bmaxv, vpre);

  // --- Wf1 1x1 cbs : WMMA GEMM, bf16 out (reuse Aimg region as ACT1) ---
  unsigned short* act1 = Aimg;
  gemm_lds_bf16<1><<<2048, blk, 0, stream>>>(vpre, wf116, sf1, bf1, nullptr, act1);

  // --- Wf2 3x3 pad1 cbs : 9-tap LDS-staged WMMA implicit GEMM, bf16 out ---
  conv3x3_lds<<<2048, blk, 0, stream>>>(act1, wf2t, sf2, bf2, act2);

  // --- Wf3 1x1 + bias : WMMA GEMM, f32 out (reuse mapv region as VFILM) ---
  float* vfilm = mapv;
  gemm_lds_bf16<0><<<2048, blk, 0, stream>>>(act2, wf316, nullptr, bf3, vfilm, nullptr);

  // --- v_add + final l2norm, NCHW output ---
  final_kernel<<<MM, blk, 0, stream>>>(vfilm, img, addw, out);
}
